// LePEAttention_68161130987693
// MI455X (gfx1250) — matrix operand order repaired and checked by reading, not compiled, not run on hardware
//
#include <hip/hip_runtime.h>

typedef __attribute__((ext_vector_type(16))) __bf16 v16bf;
typedef __attribute__((ext_vector_type(8)))  float  v8f;

namespace {
constexpr int DHEAD   = 64;    // head dim
constexpr int WSP     = 7;     // window width
constexpr int WIMG    = 56;    // image h == w
constexpr int LWIN    = 392;   // tokens per window (56*7)
constexpr int LPAD    = 416;   // padded to 13*32
constexpr int KSTRIDE = 66;    // K row stride in bf16 (33 dwords, gcd(33,64)=1)
constexpr int VSTRIDE = 422;   // Vt row stride in bf16 (211 dwords, gcd(211,64)=1)
constexpr int PSTRIDE = 34;    // P-stage row stride (17 dwords, gcd(17,64)=1)
constexpr int CCH     = 512;   // channels
constexpr int NTOK    = WIMG * WIMG; // 3136
}

__device__ __forceinline__ v16bf lds_frag16(const __bf16* p) {
  // 16 consecutive bf16 (32B), dword-aligned by construction
  union { v16bf v; unsigned u[8]; } t;
  const unsigned* q = (const unsigned*)p;
#pragma unroll
  for (int i = 0; i < 8; ++i) t.u[i] = q[i];
  return t.v;
}

__global__ __launch_bounds__(256)
void cswin_lepe_attn(const float* __restrict__ qkv,
                     const float* __restrict__ convw,
                     const float* __restrict__ convb,
                     float* __restrict__ out)
{
  extern __shared__ char smem_raw[];
  __bf16* Ksh = (__bf16*)smem_raw;            // [LPAD][KSTRIDE]  row-major K
  __bf16* Vsh = Ksh + LPAD * KSTRIDE;         // [DHEAD][VSTRIDE] transposed V
  __bf16* Psh = Vsh + DHEAD * VSTRIDE;        // [8 waves][16][PSTRIDE] P stage

  const int hd   = blockIdx.x & 7;            // head
  const int widx = blockIdx.x >> 3;           // window index 0..127
  const int wi   = widx & 7;                  // window column
  const int bi   = widx >> 3;                 // batch

  const int tid     = threadIdx.x;
  const int lane    = tid & 31;
  const int wv      = tid >> 5;
  const int halfsel = lane >> 4;              // 0: lanes 0-15, 1: lanes 16-31
  const int lan15   = lane & 15;

  const size_t plane  = (size_t)NTOK * CCH;   // b-stride inside one of q/k/v
  const float* qg = qkv;
  const float* kg = qkv + 16 * plane;
  const float* vg = qkv + 32 * plane;
  const size_t basebc = (size_t)bi * plane + (size_t)hd * DHEAD;

  __builtin_prefetch(qg + basebc, 0, 0);      // global_prefetch_b8

  // ---------- Phase 1: stage K (row-major) and V (transposed) as bf16 ----------
  for (int e = tid; e < LPAD * (DHEAD / 4); e += 256) {
    const int l  = e >> 4;
    const int d4 = (e & 15) << 2;
    float4 kq = {0.f, 0.f, 0.f, 0.f}, vq = {0.f, 0.f, 0.f, 0.f};
    if (l < LWIN) {
      const int tok = (l / WSP) * WIMG + wi * WSP + (l % WSP);
      const size_t off = basebc + (size_t)tok * CCH + d4;
      kq = *(const float4*)(kg + off);
      vq = *(const float4*)(vg + off);
    }
    __bf16* kd = Ksh + l * KSTRIDE + d4;
    kd[0] = (__bf16)kq.x; kd[1] = (__bf16)kq.y;
    kd[2] = (__bf16)kq.z; kd[3] = (__bf16)kq.w;
    Vsh[(d4 + 0) * VSTRIDE + l] = (__bf16)vq.x;
    Vsh[(d4 + 1) * VSTRIDE + l] = (__bf16)vq.y;
    Vsh[(d4 + 2) * VSTRIDE + l] = (__bf16)vq.z;
    Vsh[(d4 + 3) * VSTRIDE + l] = (__bf16)vq.w;
  }
  __syncthreads();

  __bf16* pst = Psh + wv * 16 * PSTRIDE;      // per-wave P staging

  // ---------- Phase 2: flash-attention over 16-row query tiles ----------
  for (int qt = wv; qt < 25; qt += 8) {
    const int qbase = qt * 16;

    // Q A-fragments (16x32 bf16; two k-slices d=0..31 and d=32..63).
    // A layout: lanes 0-15 row=lane, elems 0..15 = K 0..15;
    //           lanes 16-31 take K 16..31.
    int arow = qbase + lan15; if (arow >= LWIN) arow = LWIN - 1;
    const int atok = (arow / WSP) * WIMG + wi * WSP + (arow % WSP);
    const float* qp = qg + basebc + (size_t)atok * CCH;
    const int koff = halfsel * 16;
    v16bf qa0, qa1;
#pragma unroll
    for (int i = 0; i < 8; ++i) {
      qa0[i]     = (__bf16)(qp[koff + i]      * 0.125f);
      qa0[i + 8] = (__bf16)(qp[8 + koff + i] * 0.125f);
      qa1[i]     = (__bf16)(qp[32 + koff + i] * 0.125f);
      qa1[i + 8] = (__bf16)(qp[40 + koff + i] * 0.125f);
    }

    v8f m, lp;                                // lp: per-lane UNreduced partial sums
    v8f acc0 = {}, acc1 = {}, acc2 = {}, acc3 = {};
#pragma unroll
    for (int r = 0; r < 8; ++r) { m[r] = -1e30f; lp[r] = 0.f; }

#pragma unroll 1
    for (int ks = 0; ks < 13; ++ks) {
      const int kb = ks * 32;
      // --- S = Q * K^T for keys [kb, kb+32): B elems = 16 consecutive d per lane
      const int doff = halfsel * 16;
      const v16bf bk00 = lds_frag16(Ksh + (kb      + lan15) * KSTRIDE + doff);
      const v16bf bk01 = lds_frag16(Ksh + (kb      + lan15) * KSTRIDE + 32 + doff);
      const v16bf bk10 = lds_frag16(Ksh + (kb + 16 + lan15) * KSTRIDE + doff);
      const v16bf bk11 = lds_frag16(Ksh + (kb + 16 + lan15) * KSTRIDE + 32 + doff);

      v8f s0 = {}, s1 = {};
      s0 = __builtin_amdgcn_wmma_f32_16x16x32_bf16(false, qa0, false, bk00, (short)0, s0, false, false);
      s0 = __builtin_amdgcn_wmma_f32_16x16x32_bf16(false, qa1, false, bk01, (short)0, s0, false, false);
      s1 = __builtin_amdgcn_wmma_f32_16x16x32_bf16(false, qa0, false, bk10, (short)0, s1, false, false);
      s1 = __builtin_amdgcn_wmma_f32_16x16x32_bf16(false, qa1, false, bk11, (short)0, s1, false, false);

      const bool ok0 = (kb      + lan15) < LWIN;
      const bool ok1 = (kb + 16 + lan15) < LWIN;

      // --- online softmax; C layout: VGPR r -> rows r / r+8 per lane half.
      // Row-max needs a per-step cross-lane reduce; the row-SUM is kept as a
      // per-lane partial (linear under the alpha rescale) and reduced once at
      // the end of the tile -> halves the shuffle traffic in the hot loop.
#pragma unroll
      for (int r = 0; r < 8; ++r) {
        const float sv0 = ok0 ? s0[r] : -1e30f;
        const float sv1 = ok1 ? s1[r] : -1e30f;
        float mt = fmaxf(sv0, sv1);
        mt = fmaxf(mt, __shfl_xor(mt, 1, 32));
        mt = fmaxf(mt, __shfl_xor(mt, 2, 32));
        mt = fmaxf(mt, __shfl_xor(mt, 4, 32));
        mt = fmaxf(mt, __shfl_xor(mt, 8, 32));
        const float mnew  = fmaxf(m[r], mt);
        const float alpha = __expf(m[r] - mnew);
        m[r] = mnew;
        const float p0 = __expf(sv0 - mnew);
        const float p1 = __expf(sv1 - mnew);
        lp[r] = lp[r] * alpha + (p0 + p1);
        acc0[r] *= alpha; acc1[r] *= alpha; acc2[r] *= alpha; acc3[r] *= alpha;
        const int prow = r + halfsel * 8;
        pst[prow * PSTRIDE + lan15]      = (__bf16)p0;
        pst[prow * PSTRIDE + 16 + lan15] = (__bf16)p1;
      }

      // --- C-layout -> A-layout via per-wave LDS bounce (DS ops are in-order)
      v16bf pa;
      {
        const __bf16* pr = pst + lan15 * PSTRIDE;
        const int pk = halfsel * 16;
#pragma unroll
        for (int i = 0; i < 8; ++i) {
          pa[i]     = pr[pk + i];
          pa[i + 8] = pr[8 + pk + i];
        }
      }

      // --- O += P * V ; B elems = 16 consecutive keys per lane from Vt
      const int voff = kb + halfsel * 16;
      const v16bf bv0 = lds_frag16(Vsh + (0 * 16 + lan15) * VSTRIDE + voff);
      const v16bf bv1 = lds_frag16(Vsh + (1 * 16 + lan15) * VSTRIDE + voff);
      const v16bf bv2 = lds_frag16(Vsh + (2 * 16 + lan15) * VSTRIDE + voff);
      const v16bf bv3 = lds_frag16(Vsh + (3 * 16 + lan15) * VSTRIDE + voff);
      acc0 = __builtin_amdgcn_wmma_f32_16x16x32_bf16(false, pa, false, bv0, (short)0, acc0, false, false);
      acc1 = __builtin_amdgcn_wmma_f32_16x16x32_bf16(false, pa, false, bv1, (short)0, acc1, false, false);
      acc2 = __builtin_amdgcn_wmma_f32_16x16x32_bf16(false, pa, false, bv2, (short)0, acc2, false, false);
      acc3 = __builtin_amdgcn_wmma_f32_16x16x32_bf16(false, pa, false, bv3, (short)0, acc3, false, false);
    }

    // --- final cross-lane reduction of the deferred row sums (once per tile)
    v8f lsum;
#pragma unroll
    for (int r = 0; r < 8; ++r) {
      float s = lp[r];
      s += __shfl_xor(s, 1, 32);
      s += __shfl_xor(s, 2, 32);
      s += __shfl_xor(s, 4, 32);
      s += __shfl_xor(s, 8, 32);
      lsum[r] = s;
    }

    // ---------- Epilogue: normalize, add LePE (depthwise 3x3 from Vt), store ----------
    const int ch = hd * DHEAD;
    const v8f accs[4] = {acc0, acc1, acc2, acc3};
#pragma unroll
    for (int nt = 0; nt < 4; ++nt) {
      const int d = nt * 16 + lan15;
      float wr[9];
#pragma unroll
      for (int t = 0; t < 9; ++t) wr[t] = convw[(ch + d) * 9 + t];
      const float bias = convb[ch + d];
#pragma unroll
      for (int r = 0; r < 8; ++r) {
        const int row = qbase + r + halfsel * 8;
        if (row < LWIN) {
          const int ys = row / WSP, xs = row % WSP;
          float lep = bias;
#pragma unroll
          for (int dy = 0; dy < 3; ++dy) {
#pragma unroll
            for (int dx = 0; dx < 3; ++dx) {
              const int ny = ys + dy - 1, nx = xs + dx - 1;
              if (ny >= 0 && ny < WIMG && nx >= 0 && nx < WSP)
                lep += (float)Vsh[d * VSTRIDE + ny * WSP + nx] * wr[dy * 3 + dx];
            }
          }
          const float o = accs[nt][r] / lsum[r] + lep;
          const int tok = ys * WIMG + wi * WSP + xs;
          out[((size_t)bi * NTOK + tok) * CCH + ch + d] = o;
        }
      }
    }
  }
}

extern "C" void kernel_launch(void* const* d_in, const int* in_sizes, int n_in,
                              void* d_out, int out_size, void* d_ws, size_t ws_size,
                              hipStream_t stream) {
  (void)in_sizes; (void)n_in; (void)d_ws; (void)ws_size; (void)out_size;
  const float* qkv   = (const float*)d_in[0];
  const float* convw = (const float*)d_in[1];
  const float* convb = (const float*)d_in[2];
  // d_in[3] = h (56), d_in[4] = w (56): geometry compiled in.
  float* out = (float*)d_out;

  const size_t smem = (size_t)(LPAD * KSTRIDE + DHEAD * VSTRIDE + 8 * 16 * PSTRIDE)
                      * sizeof(__bf16); // ~117.6 KB -> 2 workgroups / WGP (320 KB LDS)
  dim3 grid(1024);   // 128 windows * 8 heads
  dim3 block(256);   // 8 wave32
  cswin_lepe_attn<<<grid, block, smem, stream>>>(qkv, convw, convb, out);
}
